// TppNet_13657996001733
// MI455X (gfx1250) — matrix operand order repaired and checked by reading, not compiled, not run on hardware
//
#include <hip/hip_runtime.h>
#include <hip/hip_bf16.h>

typedef __attribute__((ext_vector_type(16))) _Float16 v16h;
typedef __attribute__((ext_vector_type(8)))  _Float16 v8h;
typedef __attribute__((ext_vector_type(8)))  float    v8f;

union FragAB {
    v16h v;
    v8h  h8[2];
};

constexpr int BATCH = 32;
constexpr int NPTS  = 1000;
constexpr int KNBR  = 8;
constexpr int PPAIR = NPTS * (NPTS - 1) / 2;   // 499500

// ---------------------------------------------------------------------------
// Async global -> LDS 16B copy (ASYNCcnt path, CDNA5 ISA). Inline asm
// because the clang builtin arity differs across toolchains.
// ---------------------------------------------------------------------------
__device__ __forceinline__ void async_g2l_b128(void* lds, const void* gaddr) {
    unsigned l = (unsigned)(unsigned long long)lds;   // LDS aperture: addr[31:0]
    asm volatile("global_load_async_to_lds_b128 %0, %1, off"
                 :: "v"(l), "v"(gaddr)
                 : "memory");
}
__device__ __forceinline__ void wait_async0() {
    asm volatile("s_wait_asynccnt 0" ::: "memory");
}

// ---------------------------------------------------------------------------
// Pack an (Krows x N) fp32 weight into WMMA-B fragment order, fp16:
// dst[((nt*ksteps + kt)*32 + lane)*16 + h] = W[kt*32 + 16*(h>>3) + 8*(lane>>4) + (h&7)][nt*16 + (lane&15)]
// If dinsub>0, fold the EdgeConv input transform: rows k<dinsub become
// (w_top - w_bot), rows dinsub..2*dinsub keep w_bot, so A can be [xi, xj].
// Rows k >= Krows are zero (K padding).
// ---------------------------------------------------------------------------
__global__ void packw_kernel(const float* __restrict__ src, _Float16* __restrict__ dst,
                             int Krows, int N, int dinsub, int ksteps, int total) {
    int i = blockIdx.x * 256 + threadIdx.x;
    if (i >= total) return;
    int per = ksteps * 512;
    int nt  = i / per;
    int r   = i - nt * per;
    int kt  = r >> 9;
    int rr  = r & 511;
    int ln  = rr >> 4, hh = rr & 15;
    int col = nt * 16 + (ln & 15);
    int k   = kt * 32 + ((hh >> 3) << 4) + ((ln >> 4) << 3) + (hh & 7);
    float v = 0.f;
    if (k < Krows) {
        v = src[k * N + col];
        if (dinsub && k < dinsub) v -= src[(k + dinsub) * N + col];
    }
    dst[i] = (_Float16)v;
}

// ---------------------------------------------------------------------------
// kNN (K=8, includes self, like jax top_k on -dist).
// ---------------------------------------------------------------------------
template<int D, typename XT>
__global__ void knn_kernel(const XT* __restrict__ x, int xstride, int* __restrict__ idx) {
    constexpr int PTSB = 100;
    __shared__ float xloc[PTSB * D];
    const int b = blockIdx.y;
    const int pbase = blockIdx.x * PTSB;
    const int t = threadIdx.x;                      // 128

    for (int q = t; q < PTSB * D; q += 128) {
        int r = q / D, d = q - r * D;
        xloc[q] = (float)x[(size_t)(b * NPTS + pbase + r) * xstride + d];
    }
    __syncthreads();

    if (t < PTSB) {
        float bd[KNBR]; int bi[KNBR];
        #pragma unroll
        for (int s = 0; s < KNBR; ++s) { bd[s] = 3.4e38f; bi[s] = 0; }
        const float* xm = &xloc[t * D];
        for (int j = 0; j < NPTS; ++j) {
            const XT* xj = &x[(size_t)(b * NPTS + j) * xstride];
            float d2 = 0.f;
            #pragma unroll 8
            for (int d = 0; d < D; ++d) { float df = xm[d] - (float)xj[d]; d2 += df * df; }
            if (d2 < bd[KNBR - 1]) {
                bd[KNBR - 1] = d2; bi[KNBR - 1] = j;
                #pragma unroll
                for (int s = KNBR - 1; s > 0; --s) {
                    if (bd[s] < bd[s - 1]) {
                        float td = bd[s]; bd[s] = bd[s - 1]; bd[s - 1] = td;
                        int   ti = bi[s]; bi[s] = bi[s - 1]; bi[s - 1] = ti;
                    }
                }
            }
        }
        const int p = pbase + t;
        #pragma unroll
        for (int s = 0; s < KNBR; ++s) idx[(size_t)(b * NPTS + p) * KNBR + s] = bi[s];
    }
}

// ---------------------------------------------------------------------------
// EdgeConv with A = [xi, xj] (transform folded into packed w1).
// Block = 128 threads = 4 waves; 64 edge rows (8 points). f16 in/out features.
// Edge rows + weight tiles arrive via async global->LDS DMA where aligned.
// ---------------------------------------------------------------------------
template<typename XT, int DIN, int CH, int COUT>
__global__ void edgeconv_kernel(const XT* __restrict__ x, int xstride,
                                const int* __restrict__ idx,
                                const _Float16* __restrict__ w1p, const float* __restrict__ b1,
                                const _Float16* __restrict__ w2p, const float* __restrict__ b2,
                                _Float16* __restrict__ out, int ostride, int ocol0) {
    constexpr int KE       = 2 * DIN;
    constexpr int KE_STEPS = (KE + 31) / 32;
    constexpr int KE_PAD   = KE_STEPS * 32;
    constexpr int CH_STEPS = (CH + 31) / 32;
    constexpr int CH_PAD   = CH_STEPS * 32;
    constexpr int WKS      = (KE_STEPS > CH_STEPS) ? KE_STEPS : CH_STEPS;
    constexpr int ROWS     = 64;
    constexpr bool ASYNCE  = (sizeof(XT) == 2) && (DIN % 8 == 0);

    __shared__ _Float16 eL[ROWS * KE_PAD] __attribute__((aligned(16)));
    __shared__ _Float16 hL[ROWS * CH_PAD] __attribute__((aligned(16)));
    __shared__ _Float16 wst[WKS * 512]    __attribute__((aligned(16)));
    __shared__ int nidx[ROWS];

    const int tid = threadIdx.x;          // 128
    const int wv  = tid >> 5;
    const int ln  = tid & 31;
    const int lm  = ln & 15;
    const int lhi = ln >> 4;
    const int blk = blockIdx.x;
    const int b     = blk / (NPTS / 8);
    const int pbase = (blk % (NPTS / 8)) * 8;
    const int mbase = wv * 16;

    if (tid < ROWS) {
        int p = pbase + (tid >> 3);
        nidx[tid] = idx[(size_t)(b * NPTS + p) * KNBR + (tid & 7)];
    }
    if (KE < KE_PAD) for (int q = tid; q < ROWS * KE_PAD; q += 128) eL[q] = (_Float16)0.f;
    if (CH < CH_PAD) for (int q = tid; q < ROWS * CH_PAD; q += 128) hL[q] = (_Float16)0.f;
    __syncthreads();

    // ---- gather e = [xi | xj] -----------------------------------------
    if constexpr (ASYNCE) {
        constexpr int CPR  = DIN / 4;      // 16B chunks per edge row (2*DIN halves)
        constexpr int HALF = CPR / 2;
        for (int q = tid; q < ROWS * CPR; q += 128) {
            int r = q / CPR, c = q - r * CPR;
            int srow = (c < HALF) ? (pbase + (r >> 3)) : nidx[r];
            int coff = (c < HALF) ? c : (c - HALF);
            const XT* gp = x + (size_t)(b * NPTS + srow) * xstride + coff * 8;
            async_g2l_b128(&eL[r * KE_PAD + c * 8], gp);
        }
        wait_async0();
    } else {
        for (int q = tid; q < ROWS * DIN; q += 128) {
            int r = q / DIN, d = q - r * DIN;
            int p = pbase + (r >> 3);
            int j = nidx[r];
            eL[r * KE_PAD + d]       = (_Float16)(float)x[(size_t)(b * NPTS + p) * xstride + d];
            eL[r * KE_PAD + DIN + d] = (_Float16)(float)x[(size_t)(b * NPTS + j) * xstride + d];
        }
    }
    __syncthreads();

    // ---- GEMM1: e(64 x KE_PAD) @ w1'(KE x CH), +bias, ReLU -> hL -------
    for (int nt = 0; nt < CH / 16; ++nt) {
        for (int q = tid; q < KE_STEPS * 64; q += 128)
            async_g2l_b128(&wst[q * 8], w1p + (size_t)nt * KE_STEPS * 512 + q * 8);
        wait_async0();
        __syncthreads();
        v8f c = {};
        #pragma unroll
        for (int kt = 0; kt < KE_STEPS; ++kt) {
            FragAB a, bf;
            const _Float16* ap = &eL[(mbase + lm) * KE_PAD + kt * 32 + lhi * 8];
            a.h8[0] = *(const v8h*)ap;
            a.h8[1] = *(const v8h*)(ap + 16);
            const _Float16* bp = &wst[(kt * 32 + ln) * 16];
            bf.h8[0] = *(const v8h*)bp;
            bf.h8[1] = *(const v8h*)(bp + 8);
            c = __builtin_amdgcn_wmma_f32_16x16x32_f16(false, a.v, false, bf.v,
                                                       (short)0, c, false, false);
        }
        int col = nt * 16 + lm;
        float bs = b1[col];
        #pragma unroll
        for (int r = 0; r < 8; ++r) {
            float v = c[r] + bs;
            v = v > 0.f ? v : 0.f;
            hL[(mbase + r + (lhi << 3)) * CH_PAD + col] = (_Float16)v;
        }
        __syncthreads();
    }

    // ---- GEMM2: h @ w2, +bias, max over 8 neighbor rows ----------------
    for (int nt = 0; nt < COUT / 16; ++nt) {
        for (int q = tid; q < CH_STEPS * 64; q += 128)
            async_g2l_b128(&wst[q * 8], w2p + (size_t)nt * CH_STEPS * 512 + q * 8);
        wait_async0();
        __syncthreads();
        v8f c = {};
        #pragma unroll
        for (int kt = 0; kt < CH_STEPS; ++kt) {
            FragAB a, bf;
            const _Float16* ap = &hL[(mbase + lm) * CH_PAD + kt * 32 + lhi * 8];
            a.h8[0] = *(const v8h*)ap;
            a.h8[1] = *(const v8h*)(ap + 16);
            const _Float16* bp = &wst[(kt * 32 + ln) * 16];
            bf.h8[0] = *(const v8h*)bp;
            bf.h8[1] = *(const v8h*)(bp + 8);
            c = __builtin_amdgcn_wmma_f32_16x16x32_f16(false, a.v, false, bf.v,
                                                       (short)0, c, false, false);
        }
        int col = nt * 16 + lm;
        float bs = b2[col];
        float vm = -3.4e38f;
        #pragma unroll
        for (int r = 0; r < 8; ++r) vm = fmaxf(vm, c[r] + bs);
        int p = pbase + (wv << 1) + lhi;
        out[(size_t)(b * NPTS + p) * ostride + ocol0 + col] = (_Float16)vm;
        __syncthreads();
    }
}

// ---------------------------------------------------------------------------
// Shared MLP over all rows: f16 x(rows x KIN) -> relu CH -> f32 (rows x COUT).
// Block = 64 threads = 2 waves, 32 rows/block. KIN, CH multiples of 32.
// ---------------------------------------------------------------------------
template<int KIN, int CH, int COUT>
__global__ void mlp2_rows_kernel(const _Float16* __restrict__ x, int xstride,
                                 const _Float16* __restrict__ w1p, const float* __restrict__ b1,
                                 const _Float16* __restrict__ w2p, const float* __restrict__ b2,
                                 float* __restrict__ out, int ostride) {
    constexpr int K_STEPS  = KIN / 32;
    constexpr int CH_STEPS = CH / 32;
    constexpr int WKS      = (K_STEPS > CH_STEPS) ? K_STEPS : CH_STEPS;
    constexpr int ROWS     = 32;

    __shared__ _Float16 eL[ROWS * KIN] __attribute__((aligned(16)));
    __shared__ _Float16 hL[ROWS * CH]  __attribute__((aligned(16)));
    __shared__ _Float16 wst[WKS * 512] __attribute__((aligned(16)));

    const int tid = threadIdx.x;          // 64
    const int wv  = tid >> 5;
    const int ln  = tid & 31;
    const int lm  = ln & 15;
    const int lhi = ln >> 4;
    const int rowbase = blockIdx.x * ROWS;
    const int mbase   = wv * 16;

    constexpr int RCPR = KIN / 8;         // 16B chunks per input row
    for (int q = tid; q < ROWS * RCPR; q += 64) {
        int r = q / RCPR, c = q - r * RCPR;
        async_g2l_b128(&eL[r * KIN + c * 8],
                       x + (size_t)(rowbase + r) * xstride + c * 8);
    }
    wait_async0();
    __syncthreads();

    for (int nt = 0; nt < CH / 16; ++nt) {
        for (int q = tid; q < K_STEPS * 64; q += 64)
            async_g2l_b128(&wst[q * 8], w1p + (size_t)nt * K_STEPS * 512 + q * 8);
        wait_async0();
        __syncthreads();
        v8f c = {};
        for (int kt = 0; kt < K_STEPS; ++kt) {
            FragAB a, bf;
            const _Float16* ap = &eL[(mbase + lm) * KIN + kt * 32 + lhi * 8];
            a.h8[0] = *(const v8h*)ap;
            a.h8[1] = *(const v8h*)(ap + 16);
            const _Float16* bp = &wst[(kt * 32 + ln) * 16];
            bf.h8[0] = *(const v8h*)bp;
            bf.h8[1] = *(const v8h*)(bp + 8);
            c = __builtin_amdgcn_wmma_f32_16x16x32_f16(false, a.v, false, bf.v,
                                                       (short)0, c, false, false);
        }
        int col = nt * 16 + lm;
        float bs = b1[col];
        #pragma unroll
        for (int r = 0; r < 8; ++r) {
            float v = c[r] + bs;
            v = v > 0.f ? v : 0.f;
            hL[(mbase + r + (lhi << 3)) * CH + col] = (_Float16)v;
        }
        __syncthreads();
    }

    for (int nt = 0; nt < COUT / 16; ++nt) {
        for (int q = tid; q < CH_STEPS * 64; q += 64)
            async_g2l_b128(&wst[q * 8], w2p + (size_t)nt * CH_STEPS * 512 + q * 8);
        wait_async0();
        __syncthreads();
        v8f c = {};
        #pragma unroll
        for (int kt = 0; kt < CH_STEPS; ++kt) {
            FragAB a, bf;
            const _Float16* ap = &hL[(mbase + lm) * CH + kt * 32 + lhi * 8];
            a.h8[0] = *(const v8h*)ap;
            a.h8[1] = *(const v8h*)(ap + 16);
            const _Float16* bp = &wst[(kt * 32 + ln) * 16];
            bf.h8[0] = *(const v8h*)bp;
            bf.h8[1] = *(const v8h*)(bp + 8);
            c = __builtin_amdgcn_wmma_f32_16x16x32_f16(false, a.v, false, bf.v,
                                                       (short)0, c, false, false);
        }
        int col = nt * 16 + lm;
        float bs = b2[col];
        #pragma unroll
        for (int r = 0; r < 8; ++r)
            out[(size_t)(rowbase + mbase + r + (lhi << 3)) * ostride + col] = c[r] + bs;
        __syncthreads();
    }
}

// ---------------------------------------------------------------------------
__global__ void rowmax_kernel(const float* __restrict__ s, float* __restrict__ g) {
    int b = blockIdx.x;
    int c = threadIdx.x;                  // 32
    float m = -3.4e38f;
    for (int i = 0; i < NPTS; ++i) m = fmaxf(m, s[(size_t)(b * NPTS + i) * 32 + c]);
    g[b * 32 + c] = m;
}

__global__ void head1_kernel(const float* __restrict__ g, const float* __restrict__ w1,
                             const float* __restrict__ b1, float* __restrict__ gact) {
    int t = threadIdx.x;                  // 512 = 32*16
    int b = t >> 4, c = t & 15;
    float s = b1[c];
    #pragma unroll
    for (int j = 0; j < 32; ++j) s += g[b * 32 + j] * w1[j * 16 + c];
    gact[b * 16 + c] = s > 0.f ? s : 0.f;
}

__global__ void head2_kernel(const float* __restrict__ gact, const float* __restrict__ w2,
                             const float* __restrict__ b2, float* __restrict__ out) {
    __shared__ float ga[16];
    int b = blockIdx.y;
    if (threadIdx.x < 16) ga[threadIdx.x] = gact[b * 16 + threadIdx.x];
    __syncthreads();
    int p = blockIdx.x * 256 + threadIdx.x;
    if (p < PPAIR) {
        float s = b2[p];
        #pragma unroll
        for (int j = 0; j < 16; ++j) s += ga[j] * w2[(size_t)j * PPAIR + p];
        out[(size_t)b * PPAIR + p] = 1.f / (1.f + __expf(-s));
        out[(size_t)BATCH * PPAIR + (size_t)b * PPAIR + p] = s;
    }
}

// ---------------------------------------------------------------------------
extern "C" void kernel_launch(void* const* d_in, const int* in_sizes, int n_in,
                              void* d_out, int out_size, void* d_ws, size_t ws_size,
                              hipStream_t stream) {
    (void)in_sizes; (void)n_in; (void)out_size; (void)ws_size;
    const float* pos  = (const float*)d_in[0];
    const float* c1w1 = (const float*)d_in[1];
    const float* c1b1 = (const float*)d_in[2];
    const float* c1w2 = (const float*)d_in[3];
    const float* c1b2 = (const float*)d_in[4];
    const float* c2w1 = (const float*)d_in[5];
    const float* c2b1 = (const float*)d_in[6];
    const float* c2w2 = (const float*)d_in[7];
    const float* c2b2 = (const float*)d_in[8];
    const float* c3w1 = (const float*)d_in[9];
    const float* c3b1 = (const float*)d_in[10];
    const float* c3w2 = (const float*)d_in[11];
    const float* c3b2 = (const float*)d_in[12];
    const float* smw1 = (const float*)d_in[13];
    const float* smb1 = (const float*)d_in[14];
    const float* smw2 = (const float*)d_in[15];
    const float* smb2 = (const float*)d_in[16];
    const float* hw1  = (const float*)d_in[17];
    const float* hb1  = (const float*)d_in[18];
    const float* hw2  = (const float*)d_in[19];
    const float* hb2  = (const float*)d_in[20];

    char* ws = (char*)d_ws;
    size_t off = 0;
    auto take = [&](size_t bytes) {
        char* p = ws + off;
        off = (off + bytes + 255) & ~(size_t)255;
        return p;
    };

    _Float16* xcat   = (_Float16*)take((size_t)BATCH * NPTS * 672 * sizeof(_Float16));
    int*      idxbuf = (int*)take((size_t)BATCH * NPTS * KNBR * sizeof(int));
    float*    smout  = (float*)take((size_t)BATCH * NPTS * 32 * sizeof(float));
    float*    g      = (float*)take((size_t)BATCH * 32 * sizeof(float));
    float*    gact   = (float*)take((size_t)BATCH * 16 * sizeof(float));
    // packed fragment-order weights: ntiles * ksteps * 512 halves each
    _Float16* pc1w1 = (_Float16*)take((size_t)1  * 1  * 512 * 2);
    _Float16* pc1w2 = (_Float16*)take((size_t)2  * 1  * 512 * 2);
    _Float16* pc2w1 = (_Float16*)take((size_t)4  * 2  * 512 * 2);
    _Float16* pc2w2 = (_Float16*)take((size_t)8  * 2  * 512 * 2);
    _Float16* pc3w1 = (_Float16*)take((size_t)16 * 8  * 512 * 2);
    _Float16* pc3w2 = (_Float16*)take((size_t)32 * 8  * 512 * 2);
    _Float16* psmw1 = (_Float16*)take((size_t)16 * 21 * 512 * 2);
    _Float16* psmw2 = (_Float16*)take((size_t)2  * 8  * 512 * 2);

    auto pk = [&](const float* s, _Float16* d, int K, int N, int dinsub, int ksteps, int ntiles) {
        int total = ntiles * ksteps * 512;
        packw_kernel<<<(total + 255) / 256, 256, 0, stream>>>(s, d, K, N, dinsub, ksteps, total);
    };
    pk(c1w1, pc1w1, 6,   16,  3,   1,  1);
    pk(c1w2, pc1w2, 16,  32,  0,   1,  2);
    pk(c2w1, pc2w1, 64,  64,  32,  2,  4);
    pk(c2w2, pc2w2, 64,  128, 0,   2,  8);
    pk(c3w1, pc3w1, 256, 256, 128, 8,  16);
    pk(c3w2, pc3w2, 256, 512, 0,   8,  32);
    pk(smw1, psmw1, 672, 256, 0,   21, 16);
    pk(smw2, psmw2, 256, 32,  0,   8,  2);

    dim3 knng(NPTS / 100, BATCH);
    const int ecg = BATCH * (NPTS / 8);

    // EdgeConv 1 (pos f32, D=3) -> xcat cols [0,32)
    knn_kernel<3, float><<<knng, 128, 0, stream>>>(pos, 3, idxbuf);
    edgeconv_kernel<float, 3, 16, 32><<<ecg, 128, 0, stream>>>(
        pos, 3, idxbuf, pc1w1, c1b1, pc1w2, c1b2, xcat, 672, 0);

    // EdgeConv 2 (x1 f16, D=32) -> xcat cols [32,160)
    knn_kernel<32, _Float16><<<knng, 128, 0, stream>>>(xcat, 672, idxbuf);
    edgeconv_kernel<_Float16, 32, 64, 128><<<ecg, 128, 0, stream>>>(
        xcat, 672, idxbuf, pc2w1, c2b1, pc2w2, c2b2, xcat, 672, 32);

    // EdgeConv 3 (x2 f16, D=128) -> xcat cols [160,672)
    knn_kernel<128, _Float16><<<knng, 128, 0, stream>>>(xcat + 32, 672, idxbuf);
    edgeconv_kernel<_Float16, 128, 256, 512><<<ecg, 128, 0, stream>>>(
        xcat + 32, 672, idxbuf, pc3w1, c3b1, pc3w2, c3b2, xcat, 672, 160);

    // Shared MLP 672 -> 256 -> 32 over all B*N rows
    mlp2_rows_kernel<672, 256, 32><<<(BATCH * NPTS) / 32, 64, 0, stream>>>(
        xcat, 672, psmw1, smb1, psmw2, smb2, smout, 32);

    // Global max pool + head
    rowmax_kernel<<<BATCH, 32, 0, stream>>>(smout, g);
    head1_kernel<<<1, 512, 0, stream>>>(g, hw1, hb1, gact);
    dim3 h2g((PPAIR + 255) / 256, BATCH);
    head2_kernel<<<h2g, 256, 0, stream>>>(gact, hw2, hb2, (float*)d_out);
}
